// SA_Layer_11364483465520
// MI455X (gfx1250) — matrix operand order repaired and checked by reading, not compiled, not run on hardware
//
#include <hip/hip_runtime.h>

// ---------------------------------------------------------------------------
// SA_Layer (offset attention) for MI455X / gfx1250, wave32 + WMMA bf16 + TDM.
// B=4, C=256, C4=64, N=4096.
// All big GEMMs run on v_wmma_f32_16x16x32_bf16 (fp32 accumulate).
// attn is materialized once in bf16 (134 MB); energy is recomputed in the
// softmax second pass instead of being stored (268 MB fp32 round-trip at
// 23.3 TB/s costs far more than the extra 8.6 GF of bf16 WMMA).
// The x_r GEMM streams attn via the Tensor Data Mover into double-buffered
// per-wave LDS tiles (tensor_load_to_lds + s_wait_tensorcnt).
// ---------------------------------------------------------------------------

#define BB 4
#define CC 256
#define C4V 64
#define NN 4096
#define BN_EPS 1e-5f

typedef __attribute__((ext_vector_type(16))) __bf16 v16bf;
typedef __attribute__((ext_vector_type(8)))  float  v8f;
typedef __attribute__((ext_vector_type(8)))  unsigned int v8u;

__device__ inline unsigned short bfbits(float f) {
  unsigned u = __builtin_bit_cast(unsigned, f);
  return (unsigned short)((u + 0x7FFFu + ((u >> 16) & 1u)) >> 16);  // RNE
}
__device__ inline __bf16 f2bf(float f) {
  return __builtin_bit_cast(__bf16, bfbits(f));
}
__device__ inline float bf2f(__bf16 h) {
  unsigned u = ((unsigned)__builtin_bit_cast(unsigned short, h)) << 16;
  return __builtin_bit_cast(float, u);
}

__device__ inline v8f wmma_bf16(v16bf a, v16bf b, v8f c) {
  // (neg_a, A, neg_b, B, c_mod, C, reuse_a, reuse_b)
  return __builtin_amdgcn_wmma_f32_16x16x32_bf16(false, a, false, b,
                                                 (short)0, c, false, false);
}

// A fragment (16x32 bf16). Source: row-major [16 rows][K contiguous], stride ld
// (ld even, base 4B-aligned). ISA layout: lanes 0-15 = M rows, K-half offset 8
// for lanes 16-31; VGPRs 0-3 cover K 0..15(+8), VGPRs 4-7 cover K 16..31(+8).
__device__ inline v16bf frag_a_bf(const __bf16* p, int ld, int lane) {
  const unsigned int* row = (const unsigned int*)(p + (size_t)(lane & 15) * ld);
  int kh2 = (lane & 16) ? 4 : 0;  // K offset in dword units
  v8u t;
#pragma unroll
  for (int i = 0; i < 4; ++i) {
    t[i]     = row[kh2 + i];
    t[4 + i] = row[8 + kh2 + i];
  }
  return __builtin_bit_cast(v16bf, t);
}

// A fragment from fp32 source (converted to bf16 on load).
__device__ inline v16bf frag_a_f32(const float* p, int ld, int lane) {
  const float* row = p + (size_t)(lane & 15) * ld;
  int kh = (lane & 16) ? 8 : 0;
  v8u t;
#pragma unroll
  for (int i = 0; i < 4; ++i) {
    t[i] = (unsigned)bfbits(row[kh + 2 * i]) |
           ((unsigned)bfbits(row[kh + 2 * i + 1]) << 16);
    t[4 + i] = (unsigned)bfbits(row[16 + kh + 2 * i]) |
               ((unsigned)bfbits(row[16 + kh + 2 * i + 1]) << 16);
  }
  return __builtin_bit_cast(v16bf, t);
}

// B fragment (32x16 bf16). Source: B^T row-major [16 cols][K contiguous].
// ISA layout: lanes 0-15 = N cols; same K order as A: K 0..7,16..23 (lanes 0-15), K 8..15,24..31 (lanes 16-31).
__device__ inline v16bf frag_b_bf(const __bf16* p, int ld, int lane) {
  const unsigned int* row = (const unsigned int*)(p + (size_t)(lane & 15) * ld);
  int kb2 = (lane & 16) ? 4 : 0;
  v8u t;
#pragma unroll
  for (int j = 0; j < 8; ++j) t[j] = row[kb2 + j + (j >= 4 ? 4 : 0)];
  return __builtin_bit_cast(v16bf, t);
}

// ---------------------------------------------------------------------------
// Tensor Data Mover: 2D bf16 tile (tile_d0 contiguous elems x tile_d1 rows,
// row stride `stride` elems) from global -> LDS. D# layout per ISA 8.3/8.4.
// This toolchain exposes the 6-arg builtin:
//   (uint32x4 g0, int32x8 g1, int32x4 g2, int32x4 g3, int32x8 extra, i32 cpol)
#if __has_builtin(__builtin_amdgcn_tensor_load_to_lds) && \
    __has_builtin(__builtin_amdgcn_s_wait_tensorcnt)
#define USE_TDM 1
typedef __attribute__((ext_vector_type(4))) unsigned int v4u;
typedef __attribute__((ext_vector_type(8))) int v8i;
typedef __attribute__((ext_vector_type(4))) int v4i;

__device__ inline void tdm_load_2d_bf16(unsigned lds_off, const void* gptr,
                                        unsigned tile_d0, unsigned tile_d1,
                                        unsigned stride) {
  unsigned long long ga = (unsigned long long)(size_t)gptr;
  v4u g0;
  g0[0] = 1u;                                              // count=1, user mode
  g0[1] = lds_off;                                         // lds_addr [63:32]
  g0[2] = (unsigned)(ga & 0xFFFFFFFFu);                    // global_addr lo
  g0[3] = (unsigned)((ga >> 32) & 0x1FFFFFFu) | (2u << 30);// addr hi | type=2
  v8i g1;
  g1[0] = (int)(1u << 16);                                 // data_size=1 (2B)
  g1[1] = (int)((stride & 0xFFFFu) << 16);                 // tensor_dim0 lo16
  g1[2] = (int)(((stride >> 16) & 0xFFFFu) |               // tensor_dim0 hi16
                ((tile_d1 & 0xFFFFu) << 16));              // tensor_dim1 lo16
  g1[3] = (int)((tile_d0 & 0xFFFFu) << 16);                // tile_dim0
  g1[4] = (int)(tile_d1 & 0xFFFFu);                        // tile_dim1
  g1[5] = (int)stride;                                     // dim0_stride lo32
  g1[6] = 0;
  g1[7] = 0;
  v4i z4 = {0, 0, 0, 0};
  v8i z8 = {0, 0, 0, 0, 0, 0, 0, 0};
  __builtin_amdgcn_tensor_load_to_lds(g0, g1, z4, z4, z8, 0);
}
#else
#define USE_TDM 0
#endif

// ---------------------------------------------------------------------------
// K0: zero the column-sum accumulator
__global__ void k_zero(float* p, int n) {
  int i = blockIdx.x * 256 + threadIdx.x;
  if (i < n) p[i] = 0.f;
}

// ---------------------------------------------------------------------------
// K1: qkT[b][n][o] = sum_c wq[o][c] * x[b][c][n]   (o < 64), bf16 output.
// M=n, K=c, Ncol=o. wq held entirely in LDS (32 KB); x staged transposed.
__global__ void k_proj_qk(const float* __restrict__ x,
                          const float* __restrict__ wq,
                          __bf16* __restrict__ qkT) {
  __shared__ __bf16 s_wq[C4V * CC];     // [o][c] 32 KB
  __shared__ __bf16 s_xT[4][16 * 64];   // per-wave [n][c-sub] 2 KB each
  int tid = threadIdx.x, lane = tid & 31, wave = tid >> 5;
  int b = blockIdx.x >> 6;                       // / (N/64)
  int n0 = (blockIdx.x & 63) * 64 + wave * 16;
  for (int i = tid; i < C4V * CC; i += 128) s_wq[i] = f2bf(wq[i]);
  __syncthreads();
  const float* xb = x + (size_t)b * CC * NN;
  __bf16* xT = s_xT[wave];
  v8f z = {};
  v8f acc[4] = {z, z, z, z};
  for (int kg = 0; kg < 4; ++kg) {               // c groups of 64
    for (int i = lane; i < 16 * 64; i += 32) {   // wave-private stage (in-order LDS)
      int c = i >> 4, nn_ = i & 15;
      xT[nn_ * 64 + c] = f2bf(xb[(size_t)(kg * 64 + c) * NN + n0 + nn_]);
    }
#pragma unroll
    for (int kc2 = 0; kc2 < 2; ++kc2) {
      v16bf a = frag_a_bf(xT + kc2 * 32, 64, lane);
#pragma unroll
      for (int ot = 0; ot < 4; ++ot) {
        v16bf bb = frag_b_bf(s_wq + (ot * 16) * CC + kg * 64 + kc2 * 32, CC, lane);
        acc[ot] = wmma_bf16(a, bb, acc[ot]);
      }
    }
  }
  int half = (lane >> 4) & 1, col = lane & 15;
#pragma unroll
  for (int ot = 0; ot < 4; ++ot)
#pragma unroll
    for (int r = 0; r < 8; ++r) {
      int n = n0 + r + 8 * half;
      qkT[((size_t)b * NN + n) * C4V + ot * 16 + col] = f2bf(acc[ot][r]);
    }
}

// ---------------------------------------------------------------------------
// K2: v[b][o][n] = sum_c wv[o][c]*x[b][c][n] + bv[o], bf16 output, [c][m] layout.
// M=o, K=c, Ncol=n. x tile staged transposed in LDS (shared by 4 waves).
__global__ void k_proj_v(const float* __restrict__ x,
                         const float* __restrict__ wv,
                         const float* __restrict__ bv,
                         __bf16* __restrict__ vbf) {
  __shared__ __bf16 s_xT[16 * 32];   // [n][c-chunk]
  int tid = threadIdx.x, lane = tid & 31, wave = tid >> 5;
  int n0 = blockIdx.x * 16;
  int o0 = blockIdx.y * 64 + wave * 16;
  int b = blockIdx.z;
  const float* xb = x + (size_t)b * CC * NN;
  v8f acc = {};
  for (int kc = 0; kc < 8; ++kc) {
    __syncthreads();
    for (int i = tid; i < 512; i += 128) {
      int c = i >> 4, nn_ = i & 15;
      s_xT[nn_ * 32 + c] = f2bf(xb[(size_t)(kc * 32 + c) * NN + n0 + nn_]);
    }
    __syncthreads();
    v16bf a = frag_a_f32(wv + (size_t)o0 * CC + kc * 32, CC, lane);
    v16bf bb = frag_b_bf(s_xT, 32, lane);
    acc = wmma_bf16(a, bb, acc);
  }
  int half = (lane >> 4) & 1, n = n0 + (lane & 15);
#pragma unroll
  for (int r = 0; r < 8; ++r) {
    int o = o0 + r + 8 * half;
    vbf[((size_t)b * CC + o) * NN + n] = f2bf(acc[r] + bv[o]);
  }
}

// ---------------------------------------------------------------------------
// K3: energy = q·k (recomputed twice), row softmax (flash-style online pass +
// normalize/store pass), bf16 attn out, fp32 column sums via atomics.
// Each wave owns 16 query rows. A = qkT rows (q), B = qkT rows (k).
__global__ void k_attn(const __bf16* __restrict__ qkT,
                       __bf16* __restrict__ attn,
                       float* __restrict__ colsum) {
  int tid = threadIdx.x, lane = tid & 31, wave = tid >> 5;
  int b = blockIdx.y;
  int n0 = (blockIdx.x * 4 + wave) * 16;
  const __bf16* qb = qkT + (size_t)b * NN * C4V;
  v16bf a0 = frag_a_bf(qb + (size_t)n0 * C4V, C4V, lane);
  v16bf a1 = frag_a_bf(qb + (size_t)n0 * C4V + 32, C4V, lane);
  int half = (lane >> 4) & 1;

  float rmax[8], rsum[8];
#pragma unroll
  for (int r = 0; r < 8; ++r) { rmax[r] = -3.0e38f; rsum[r] = 0.f; }

  // pass 1: online row max / sum over this lane's m-stripe
  for (int mt = 0; mt < NN / 16; ++mt) {
    const __bf16* kb = qb + (size_t)(mt * 16) * C4V;
    v8f acc = {};
    acc = wmma_bf16(a0, frag_b_bf(kb, C4V, lane), acc);
    acc = wmma_bf16(a1, frag_b_bf(kb + 32, C4V, lane), acc);
#pragma unroll
    for (int r = 0; r < 8; ++r) {
      float e = acc[r];
      float mn = fmaxf(rmax[r], e);
      rsum[r] = rsum[r] * __expf(rmax[r] - mn) + __expf(e - mn);
      rmax[r] = mn;
    }
  }
  // combine the 16 lane-stripes of each row (xor 1,2,4,8 stays in 16-group)
#pragma unroll
  for (int r = 0; r < 8; ++r) {
    float m = rmax[r], s = rsum[r];
#pragma unroll
    for (int d = 1; d < 16; d <<= 1) {
      float mo = __shfl_xor(m, d, 32);
      float so = __shfl_xor(s, d, 32);
      float mn = fmaxf(m, mo);
      s = s * __expf(m - mn) + so * __expf(mo - mn);
      m = mn;
    }
    rmax[r] = m;
    rsum[r] = 1.0f / s;
  }
  // pass 2: recompute energy, write normalized attn (bf16), column partials
  for (int mt = 0; mt < NN / 16; ++mt) {
    const __bf16* kb = qb + (size_t)(mt * 16) * C4V;
    v8f acc = {};
    acc = wmma_bf16(a0, frag_b_bf(kb, C4V, lane), acc);
    acc = wmma_bf16(a1, frag_b_bf(kb + 32, C4V, lane), acc);
    int m = mt * 16 + (lane & 15);
    float cs = 0.f;
#pragma unroll
    for (int r = 0; r < 8; ++r) {
      float p = __expf(acc[r] - rmax[r]) * rsum[r];
      cs += p;
      int n = n0 + r + 8 * half;
      attn[((size_t)b * NN + n) * NN + m] = f2bf(p);
    }
    cs += __shfl_xor(cs, 16, 32);   // both halves hold the same column
    if (lane < 16) atomicAdd(&colsum[b * NN + m], cs);
  }
}

// ---------------------------------------------------------------------------
// K4: fold the offset-attention renorm into v:  vs = v / (1e-9 + colsum[m])
__global__ void k_scale_v(const __bf16* __restrict__ vbf,
                          const float* __restrict__ colsum,
                          __bf16* __restrict__ vs) {
  size_t i = (size_t)blockIdx.x * 256 + threadIdx.x;
  int m = (int)(i & (NN - 1));
  int b = (int)(i >> 20);   // / (C*N = 2^20)
  float inv = 1.0f / (1e-9f + colsum[b * NN + m]);
  vs[i] = f2bf(bf2f(vbf[i]) * inv);
}

// ---------------------------------------------------------------------------
// K5: x_r[c][n] = sum_m vs[c][m] * attn[n][m];  xs[b][n][c] = bf16(x - x_r).
// M=n, K=m (4096), Ncol=c. attn tiles (16 rows x 256 cols, 8 KB) are DMA'd by
// the Tensor Data Mover into per-wave double-buffered LDS; DMA of tile k+1
// overlaps the 8 WMMAs of tile k (TENSORcnt <= 1 pipelining).
__global__ void k_xr(const __bf16* __restrict__ attn,
                     const __bf16* __restrict__ vs,
                     const float* __restrict__ x,
                     __bf16* __restrict__ xs) {
  __shared__ __bf16 s_at[4][2][16 * 256];   // 4 waves x 2 buffers x 8 KB = 64 KB
  int tid = threadIdx.x, lane = tid & 31, wave = tid >> 5;
  int b = blockIdx.z;
  int n0 = blockIdx.x * 16;
  int c0 = blockIdx.y * 64 + wave * 16;
  const __bf16* at = attn + ((size_t)b * NN + n0) * NN;
  const __bf16* vr = vs + ((size_t)b * CC + c0) * NN;
  __bf16* buf[2] = {s_at[wave][0], s_at[wave][1]};
#if USE_TDM
  tdm_load_2d_bf16((unsigned)(size_t)buf[0], at, 256, 16, NN);
#endif
  v8f acc = {};
  for (int kg = 0; kg < 16; ++kg) {
    __bf16* cur = buf[kg & 1];
#if USE_TDM
    if (kg + 1 < 16) {
      tdm_load_2d_bf16((unsigned)(size_t)buf[(kg + 1) & 1], at + (kg + 1) * 256,
                       256, 16, NN);
      __builtin_amdgcn_s_wait_tensorcnt(1);   // previous tile landed
    } else {
      __builtin_amdgcn_s_wait_tensorcnt(0);
    }
#else
    for (int i = lane; i < 16 * 256; i += 32) {   // wave-private stage
      int row = i >> 8, col = i & 255;
      cur[i] = at[(size_t)row * NN + kg * 256 + col];
    }
#endif
    if (kg + 1 < 16)  // pull next vs stripe toward the caches
      __builtin_prefetch(vr + (size_t)(lane & 15) * NN + (kg + 1) * 256, 0, 1);
#pragma unroll
    for (int kc2 = 0; kc2 < 8; ++kc2) {
      v16bf a = frag_a_bf(cur + kc2 * 32, 256, lane);
      v16bf bb = frag_b_bf(vr + kg * 256 + kc2 * 32, NN, lane);
      acc = wmma_bf16(a, bb, acc);
    }
  }
  int half = (lane >> 4) & 1, c = c0 + (lane & 15);
#pragma unroll
  for (int r = 0; r < 8; ++r) {
    int n = n0 + r + 8 * half;
    float xv = x[((size_t)b * CC + c) * NN + n];
    xs[((size_t)b * NN + n) * CC + c] = f2bf(xv - acc[r]);
  }
}

// ---------------------------------------------------------------------------
// K6: t = wt @ (x - x_r) + bt -> BN(inference) -> relu -> + x  => d_out.
// M=o, K=c, Ncol=n. A = wt rows (fp32->bf16), B = xs rows [n][c] (contiguous K).
__global__ void k_out(const __bf16* __restrict__ xs,
                      const float* __restrict__ wt,
                      const float* __restrict__ bt,
                      const float* __restrict__ gamma,
                      const float* __restrict__ beta,
                      const float* __restrict__ rmean,
                      const float* __restrict__ rvar,
                      const float* __restrict__ x,
                      float* __restrict__ out) {
  int tid = threadIdx.x, lane = tid & 31, wave = tid >> 5;
  int b = blockIdx.z;
  int n0 = blockIdx.x * 16;
  int o0 = blockIdx.y * 64 + wave * 16;
  const __bf16* xsb = xs + ((size_t)b * NN + n0) * CC;
  v8f acc = {};
#pragma unroll
  for (int kc = 0; kc < 8; ++kc) {
    v16bf a = frag_a_f32(wt + (size_t)o0 * CC + kc * 32, CC, lane);
    v16bf bb = frag_b_bf(xsb + kc * 32, CC, lane);
    acc = wmma_bf16(a, bb, acc);
  }
  int half = (lane >> 4) & 1, n = n0 + (lane & 15);
#pragma unroll
  for (int r = 0; r < 8; ++r) {
    int o = o0 + r + 8 * half;
    float t = acc[r] + bt[o];
    float bn = (t - rmean[o]) * (gamma[o] * rsqrtf(rvar[o] + BN_EPS)) + beta[o];
    size_t idx = ((size_t)b * CC + o) * NN + n;
    out[idx] = x[idx] + fmaxf(bn, 0.f);
  }
}

// ---------------------------------------------------------------------------
// Workspace layout (bytes):
static const size_t OFF_ATTN = 0;                                     // 134 MB
static const size_t OFF_QKT  = OFF_ATTN + (size_t)BB * NN * NN * 2;   //   2 MB
static const size_t OFF_V    = OFF_QKT + (size_t)BB * NN * C4V * 2;   //   8 MB
static const size_t OFF_VS   = OFF_V   + (size_t)BB * CC * NN * 2;    //   8 MB
static const size_t OFF_XS   = OFF_VS  + (size_t)BB * CC * NN * 2;    //   8 MB
static const size_t OFF_CS   = OFF_XS  + (size_t)BB * NN * CC * 2;    //  64 KB

extern "C" void kernel_launch(void* const* d_in, const int* in_sizes, int n_in,
                              void* d_out, int out_size, void* d_ws, size_t ws_size,
                              hipStream_t stream) {
  (void)in_sizes; (void)n_in; (void)out_size; (void)ws_size;
  const float* x     = (const float*)d_in[0];
  const float* wq    = (const float*)d_in[1];
  const float* wv    = (const float*)d_in[2];
  const float* bv    = (const float*)d_in[3];
  const float* wt    = (const float*)d_in[4];
  const float* bt    = (const float*)d_in[5];
  const float* gamma = (const float*)d_in[6];
  const float* beta  = (const float*)d_in[7];
  const float* rmean = (const float*)d_in[8];
  const float* rvar  = (const float*)d_in[9];
  float* out = (float*)d_out;

  char* ws = (char*)d_ws;
  __bf16* attn   = (__bf16*)(ws + OFF_ATTN);
  __bf16* qkT    = (__bf16*)(ws + OFF_QKT);
  __bf16* vbf    = (__bf16*)(ws + OFF_V);
  __bf16* vsbuf  = (__bf16*)(ws + OFF_VS);
  __bf16* xsbuf  = (__bf16*)(ws + OFF_XS);
  float*  colsum = (float*)(ws + OFF_CS);

  k_zero<<<(BB * NN + 255) / 256, 256, 0, stream>>>(colsum, BB * NN);
  k_proj_qk<<<BB * (NN / 64), 128, 0, stream>>>(x, wq, qkT);
  k_proj_v<<<dim3(NN / 16, 4, BB), 128, 0, stream>>>(x, wv, bv, vbf);
  k_attn<<<dim3(NN / 64, BB), 128, 0, stream>>>(qkT, attn, colsum);
  k_scale_v<<<(unsigned)((size_t)BB * CC * NN / 256), 256, 0, stream>>>(vbf, colsum, vsbuf);
  k_xr<<<dim3(NN / 16, 4, BB), 128, 0, stream>>>(attn, vsbuf, x, xsbuf);
  k_out<<<dim3(NN / 16, 4, BB), 128, 0, stream>>>(xsbuf, wt, bt, gamma, beta,
                                                  rmean, rvar, x, out);
}
